// LSTM_59682865545804
// MI455X (gfx1250) — matrix operand order repaired and checked by reading, not compiled, not run on hardware
//
#include <hip/hip_runtime.h>
#include <cmath>

// ---------------------------------------------------------------------------
// LSTM forward on gfx1250: bf16 WMMA (v_wmma_f32_16x16x32_bf16), fp32 state.
// Fused per-timestep kernel: gates = h @ Wh^T + x_t @ Wx^T (K = 1024+256),
// fully-unrolled software-pipelined K-loops, then elementwise cell update.
// 512 stream-ordered launches implement the scan.
// ---------------------------------------------------------------------------

typedef __bf16 bf16_t;
typedef __attribute__((ext_vector_type(16))) __bf16 v16bf;
typedef __attribute__((ext_vector_type(8)))  __bf16 v8bf;
typedef __attribute__((ext_vector_type(8)))  float  v8f;

#define B_   128
#define T_   512
#define I_   256
#define H_   1024
#define O_   256
#define KTOT (H_ + I_)        // 1280 combined K per gate
#define LDSS (KTOT + 8)       // padded LDS row stride (breaks 64-bank conflicts)
#define STEP_BLOCK 128        // 4 waves per block

__device__ inline v16bf combine16(v8bf lo, v8bf hi) {
  union { v16bf v; v8bf h[2]; } u;
  u.h[0] = lo; u.h[1] = hi;
  return u.v;
}

__device__ inline v8f wmma_bf16(v16bf a, v16bf b, v8f c) {
  // D = A(16x32 bf16) * B(32x16 bf16) + C(16x16 f32)
  return __builtin_amdgcn_wmma_f32_16x16x32_bf16(
      /*neg_a=*/false, a, /*neg_b=*/false, b,
      /*c_mod=*/(short)0, c, /*reuse_a=*/false, /*reuse_b=*/false);
}

__device__ inline float sigmoid_fast(float x) {
  return 1.0f / (1.0f + __expf(-x));
}
__device__ inline float tanh_fast(float x) {
  // tanh(x) = 1 - 2/(exp(2x)+1); saturates correctly at +/-1
  return 1.0f - 2.0f / (__expf(2.0f * x) + 1.0f);
}

// ---------------------------------------------------------------------------
// One LSTM timestep: grid (8 m-tiles over B, 16 n-groups), block 128 = 4 waves.
// Each wave: one 16x16 tile of [B,H], all 4 gates, pipelined K-loop.
// ---------------------------------------------------------------------------
__global__ __launch_bounds__(STEP_BLOCK) void lstm_step_kernel(
    const bf16_t* __restrict__ h_in,  bf16_t* __restrict__ h_out,
    float* __restrict__ c_state,
    const bf16_t* __restrict__ xbf,                    // [B,T,I] bf16
    const bf16_t* __restrict__ Wgh, const bf16_t* __restrict__ Wih,
    const bf16_t* __restrict__ Wfh, const bf16_t* __restrict__ Woh,   // [H,H]
    const bf16_t* __restrict__ Wgx, const bf16_t* __restrict__ Wix,
    const bf16_t* __restrict__ Wfx, const bf16_t* __restrict__ Wox,   // [H,I]
    const float* __restrict__ bg, const float* __restrict__ bi,
    const float* __restrict__ bfv_, const float* __restrict__ bo,
    int t)
{
  __shared__ bf16_t sA[16 * LDSS];   // 16 batch rows x (h | x_t) = 40.25 KB

  const int tid = threadIdx.x;
  const int b0  = blockIdx.x * 16;

  // Stage A strip into LDS: cols [0,1024) = h_in row, [1024,1280) = x[b][t][:]
  for (int c8 = tid; c8 < (16 * KTOT) / 8; c8 += STEP_BLOCK) {
    int r  = c8 / (KTOT / 8);
    int kc = (c8 % (KTOT / 8)) * 8;
    const bf16_t* src = (kc < H_)
        ? (h_in + (size_t)(b0 + r) * H_ + kc)
        : (xbf + ((size_t)(b0 + r) * T_ + t) * I_ + (kc - H_));
    *(v8bf*)&sA[r * LDSS + kc] = *(const v8bf*)src;
  }
  __syncthreads();

  const int wave = tid >> 5;
  const int lane = tid & 31;
  const int j0   = (blockIdx.y * 4 + wave) * 16;   // hidden-unit tile base
  // B-fragment addressing: lane&15 = N column (= weight row), halves = K+/-16
  const int nB  = lane & 15;
  const int kbB = (lane >> 4) * 16;
  const int row = j0 + nB;
  // A-fragment addressing: lane&15 = M row, halves = K+/-16
  const int mA  = lane & 15;
  const int cA  = (lane >> 4) * 16;

  // Hoisted per-lane bases; all subsequent loads use compile-time immediate
  // offsets (full unroll) -> saddr-form loads, no per-iteration address VALU.
  const bf16_t* aBase = &sA[mA * LDSS + cA];
  const int offWh = row * H_ + kbB;   // element offset into [H,H] matrices
  const int offWx = row * I_ + kbB;   // element offset into [H,I] matrices

  auto loadA = [&](int k) -> v16bf {   // k = unified A column (0..1279)
    const bf16_t* pa = aBase + k;
    return combine16(*(const v8bf*)pa, *(const v8bf*)(pa + 8));
  };
  auto load_h = [&](int k, v16bf& a, v16bf& wg, v16bf& wi,
                    v16bf& wf, v16bf& wo) {       // k in [0,1024)
    a  = loadA(k);
    wg = *(const v16bf*)(Wgh + offWh + k);
    wi = *(const v16bf*)(Wih + offWh + k);
    wf = *(const v16bf*)(Wfh + offWh + k);
    wo = *(const v16bf*)(Woh + offWh + k);
  };
  auto load_x = [&](int k, v16bf& a, v16bf& wg, v16bf& wi,
                    v16bf& wf, v16bf& wo) {       // k in [0,256)
    a  = loadA(H_ + k);
    wg = *(const v16bf*)(Wgx + offWx + k);
    wi = *(const v16bf*)(Wix + offWx + k);
    wf = *(const v16bf*)(Wfx + offWx + k);
    wo = *(const v16bf*)(Wox + offWx + k);
  };

  v8f accg = {}, acci = {}, accf = {}, acco = {};

  // Software-pipelined, fully unrolled: next stage's loads are in flight
  // while the four WMMAs for the current stage execute.
  v16bf a_c, g_c, i_c, f_c, o_c;
  load_h(0, a_c, g_c, i_c, f_c, o_c);
#pragma unroll
  for (int kk = 0; kk < H_ / 32; ++kk) {          // 32 iterations, Wh region
    v16bf a_u = a_c, g_u = g_c, i_u = i_c, f_u = f_c, o_u = o_c;
    const int k1 = kk * 32 + 32;
    if (k1 < H_) load_h(k1, a_c, g_c, i_c, f_c, o_c);
    else         load_x(0,  a_c, g_c, i_c, f_c, o_c);   // boundary hand-off
    accg = wmma_bf16(a_u, g_u, accg);
    acci = wmma_bf16(a_u, i_u, acci);
    accf = wmma_bf16(a_u, f_u, accf);
    acco = wmma_bf16(a_u, o_u, acco);
  }
#pragma unroll
  for (int kk = 0; kk < I_ / 32; ++kk) {          // 8 iterations, Wx region
    v16bf a_u = a_c, g_u = g_c, i_u = i_c, f_u = f_c, o_u = o_c;
    const int k1 = kk * 32 + 32;
    if (k1 < I_) load_x(k1, a_c, g_c, i_c, f_c, o_c);
    accg = wmma_bf16(a_u, g_u, accg);
    acci = wmma_bf16(a_u, i_u, acci);
    accf = wmma_bf16(a_u, f_u, accf);
    acco = wmma_bf16(a_u, o_u, acco);
  }

  // Elementwise LSTM update. D layout: reg v, lane -> (M = v + 8*(lane>>4),
  // N = lane&15).
  const int   j    = j0 + nB;
  const float bgv  = bg[j], biv = bi[j], bfv = bfv_[j], bov = bo[j];
  const int   mb   = b0 + (lane >> 4) * 8;
#pragma unroll
  for (int v = 0; v < 8; ++v) {
    const int idx = (mb + v) * H_ + j;
    float g  = tanh_fast(accg[v] + bgv);
    float ig = sigmoid_fast(acci[v] + biv);
    float fg = sigmoid_fast(accf[v] + bfv);
    float og = sigmoid_fast(acco[v] + bov);
    float cn = g * ig + c_state[idx] * fg;
    c_state[idx] = cn;
    h_out[idx]   = (bf16_t)(tanh_fast(cn) * og);
  }
}

// ---------------------------------------------------------------------------
// Final projection: out[b,o] = h[b,:] . Wph[o,:] + bp[o]   (128x256, K=1024)
// grid (8, 2), block 256: wave per 16x16 tile, pipelined + fully unrolled.
// ---------------------------------------------------------------------------
__global__ __launch_bounds__(256) void proj_kernel(
    const bf16_t* __restrict__ h, const bf16_t* __restrict__ Wph,
    const float* __restrict__ bp, float* __restrict__ out)
{
  const int tid  = threadIdx.x;
  const int wave = tid >> 5, lane = tid & 31;
  const int b0   = blockIdx.x * 16;
  const int j0   = (blockIdx.y * 8 + wave) * 16;
  const int nB = lane & 15, kbB = (lane >> 4) * 16;
  const int mA = lane & 15, cA  = (lane >> 4) * 16;

  const bf16_t* aBase = h + (size_t)(b0 + mA) * H_ + cA;
  const int     offW  = (j0 + nB) * H_ + kbB;

  auto load_stage = [&](int k, v16bf& a, v16bf& b) {
    const bf16_t* pa = aBase + k;
    a = combine16(*(const v8bf*)pa, *(const v8bf*)(pa + 8));
    b = *(const v16bf*)(Wph + offW + k);
  };

  v8f acc = {};
  v16bf a_c, b_c;
  load_stage(0, a_c, b_c);
#pragma unroll
  for (int kk = 0; kk < H_ / 32; ++kk) {
    v16bf a_u = a_c, b_u = b_c;
    const int k1 = kk * 32 + 32;
    if (k1 < H_) load_stage(k1, a_c, b_c);
    acc = wmma_bf16(a_u, b_u, acc);
  }
  const int   o    = j0 + nB;
  const float bias = bp[o];
  const int   mb   = b0 + (lane >> 4) * 8;
#pragma unroll
  for (int v = 0; v < 8; ++v)
    out[(mb + v) * O_ + o] = acc[v] + bias;
}

// ---------------------------------------------------------------------------
// Helpers: fp32 -> bf16 conversion, state zero-init.
// ---------------------------------------------------------------------------
__global__ void cvt_f32_bf16(const float* __restrict__ in,
                             bf16_t* __restrict__ out, int n) {
  int i = blockIdx.x * 256 + threadIdx.x;
  if (i < n) out[i] = (bf16_t)in[i];
}

__global__ void init_state(bf16_t* h0, bf16_t* h1, float* c, int n) {
  int i = blockIdx.x * 256 + threadIdx.x;
  if (i < n) { h0[i] = (bf16_t)0.0f; h1[i] = (bf16_t)0.0f; c[i] = 0.0f; }
}

// ---------------------------------------------------------------------------
extern "C" void kernel_launch(void* const* d_in, const int* in_sizes, int n_in,
                              void* d_out, int out_size, void* d_ws, size_t ws_size,
                              hipStream_t stream) {
  (void)in_sizes; (void)n_in; (void)out_size; (void)ws_size;

  const float* x   = (const float*)d_in[0];
  const float* Wgx = (const float*)d_in[1];
  const float* Wgh = (const float*)d_in[2];
  const float* bg  = (const float*)d_in[3];
  const float* Wix = (const float*)d_in[4];
  const float* Wih = (const float*)d_in[5];
  const float* bi  = (const float*)d_in[6];
  const float* Wfx = (const float*)d_in[7];
  const float* Wfh = (const float*)d_in[8];
  const float* bf  = (const float*)d_in[9];
  const float* Wox = (const float*)d_in[10];
  const float* Woh = (const float*)d_in[11];
  const float* bo  = (const float*)d_in[12];
  const float* Wph = (const float*)d_in[13];
  const float* bp  = (const float*)d_in[14];

  char*  ws  = (char*)d_ws;
  size_t off = 0;
  auto alloc = [&](size_t bytes) -> void* {
    void* p = ws + off;
    off += (bytes + 255) & ~(size_t)255;
    return p;
  };

  bf16_t* xbf  = (bf16_t*)alloc((size_t)B_ * T_ * I_ * 2);   // 33.5 MB
  bf16_t* wghb = (bf16_t*)alloc((size_t)H_ * H_ * 2);
  bf16_t* wihb = (bf16_t*)alloc((size_t)H_ * H_ * 2);
  bf16_t* wfhb = (bf16_t*)alloc((size_t)H_ * H_ * 2);
  bf16_t* wohb = (bf16_t*)alloc((size_t)H_ * H_ * 2);
  bf16_t* wgxb = (bf16_t*)alloc((size_t)H_ * I_ * 2);
  bf16_t* wixb = (bf16_t*)alloc((size_t)H_ * I_ * 2);
  bf16_t* wfxb = (bf16_t*)alloc((size_t)H_ * I_ * 2);
  bf16_t* woxb = (bf16_t*)alloc((size_t)H_ * I_ * 2);
  bf16_t* wphb = (bf16_t*)alloc((size_t)O_ * H_ * 2);
  bf16_t* h0   = (bf16_t*)alloc((size_t)B_ * H_ * 2);
  bf16_t* h1   = (bf16_t*)alloc((size_t)B_ * H_ * 2);
  float*  cst  = (float*)alloc((size_t)B_ * H_ * 4);

  auto cvt = [&](const float* in, bf16_t* outp, int n) {
    cvt_f32_bf16<<<(n + 255) / 256, 256, 0, stream>>>(in, outp, n);
  };
  cvt(x,   xbf,  B_ * T_ * I_);
  cvt(Wgh, wghb, H_ * H_);
  cvt(Wih, wihb, H_ * H_);
  cvt(Wfh, wfhb, H_ * H_);
  cvt(Woh, wohb, H_ * H_);
  cvt(Wgx, wgxb, H_ * I_);
  cvt(Wix, wixb, H_ * I_);
  cvt(Wfx, wfxb, H_ * I_);
  cvt(Wox, woxb, H_ * I_);
  cvt(Wph, wphb, O_ * H_);
  init_state<<<(B_ * H_ + 255) / 256, 256, 0, stream>>>(h0, h1, cst, B_ * H_);

  bf16_t* hb[2] = {h0, h1};
  for (int t = 0; t < T_; ++t) {
    lstm_step_kernel<<<dim3(8, 16), STEP_BLOCK, 0, stream>>>(
        hb[t & 1], hb[(t + 1) & 1], cst, xbf,
        wghb, wihb, wfhb, wohb,
        wgxb, wixb, wfxb, woxb,
        bg, bi, bf, bo, t);
  }
  // T even -> final h lands back in hb[0]
  proj_kernel<<<dim3(8, 2), 256, 0, stream>>>(hb[0], wphb, bp, (float*)d_out);
}
